// ForceDime_86706799772326
// MI455X (gfx1250) — matrix operand order repaired and checked by reading, not compiled, not run on hardware
//
#include <hip/hip_runtime.h>
#include <hip/hip_bf16.h>
#include <stdint.h>

#define E_EDGES 262144
#define T_TRIP  1048576
#define DIM     128
#define NRBF    6
#define NSBF    42

typedef __attribute__((ext_vector_type(16))) __bf16 v16bf;
typedef __attribute__((ext_vector_type(8)))  float  v8f;

struct Frag { uint32_t u[8]; };   // 32 bytes == one 16x32 bf16 A/B fragment per lane

static __device__ __forceinline__ v16bf frag_cast(const Frag& f) {
    return __builtin_bit_cast(v16bf, f);
}
static __device__ __forceinline__ uint16_t f2bf(float f) {
    return __builtin_bit_cast(uint16_t, (__bf16)f);
}
static __device__ __forceinline__ float bf2f(uint16_t u) {
    return __builtin_bit_cast(float, ((uint32_t)u) << 16);
}
static __device__ __forceinline__ uint32_t pk2(float lo, float hi) {
    return (uint32_t)f2bf(lo) | ((uint32_t)f2bf(hi) << 16);
}
static __device__ __forceinline__ float silu_f(float x) {
    return x / (1.0f + __expf(-x));
}
static __device__ __forceinline__ v8f vzero() {
    v8f z = {0.f,0.f,0.f,0.f,0.f,0.f,0.f,0.f};
    return z;
}
#define LDS_FENCE() asm volatile("" ::: "memory")

// Stage a KxDIM fp32 weight matrix into LDS as bf16 in B-fragment order.
// Layout: u32 index p = ((c*8 + t)*32 + lane)*8 + v
//   lane: h = lane>>4, n = 16*t + (lane&15), k0 = 32*c + 16*(v>>2) + 8*h + 2*(v&3)
static __device__ void stage_wchunks(uint32_t* dst, const float* __restrict__ W,
                                     int krows, int nchunks, int tid, int nthr) {
    int total = nchunks * 2048;
    for (int p = tid; p < total; p += nthr) {
        int v    = p & 7;
        int lane = (p >> 3) & 31;
        int t    = (p >> 8) & 7;
        int c    = p >> 11;
        int h    = lane >> 4;
        int n    = t * 16 + (lane & 15);
        int k0   = c * 32 + (v >> 2) * 16 + h * 8 + (v & 3) * 2;
        float a = (k0     < krows) ? W[(size_t)k0       * DIM + n] : 0.f;
        float b = (k0 + 1 < krows) ? W[(size_t)(k0 + 1) * DIM + n] : 0.f;
        dst[p] = pk2(a, b);
    }
}

static __device__ __forceinline__ v16bf load_bfrag(const uint32_t* wl, int idx, int lane) {
    const uint4* p = (const uint4*)(wl + ((idx * 32 + lane) * 8));
    uint4 lo = p[0], hi = p[1];
    Frag f;
    f.u[0]=lo.x; f.u[1]=lo.y; f.u[2]=lo.z; f.u[3]=lo.w;
    f.u[4]=hi.x; f.u[5]=hi.y; f.u[6]=hi.z; f.u[7]=hi.w;
    return frag_cast(f);
}

// Two row-tiles share every B fragment (halves LDS traffic per WMMA);
// B fragments are software-pipelined one step ahead of their WMMAs.
static __device__ __forceinline__ void gemm_layer2(const Frag* A0, const Frag* A1,
                                                   const uint32_t* wl, int lane,
                                                   v8f* acc0, v8f* acc1) {
    v16bf b = load_bfrag(wl, 0, lane);
#pragma unroll
    for (int c = 0; c < 4; ++c) {
        v16bf a0 = frag_cast(A0[c]);
        v16bf a1 = frag_cast(A1[c]);
#pragma unroll
        for (int t = 0; t < 8; ++t) {
            int idx = c * 8 + t;
            v16bf bn = (idx < 31) ? load_bfrag(wl, idx + 1, lane) : b;
            acc0[t] = __builtin_amdgcn_wmma_f32_16x16x32_bf16(
                false, a0, false, b, (short)0, acc0[t], false, false);
            acc1[t] = __builtin_amdgcn_wmma_f32_16x16x32_bf16(
                false, a1, false, b, (short)0, acc1[t], false, false);
            b = bn;
        }
    }
}

// C layout (lane = fixed feature, VGPR r = row) -> row-major bf16 staging [row][feat]
static __device__ __forceinline__ void act_store(const v8f* acc, const float* bias,
                                                 __bf16* stg, int lane, bool doAct) {
    int nl = lane & 15, h = lane >> 4;
#pragma unroll
    for (int t = 0; t < 8; ++t) {
        int F = t * 16 + nl;
        float bv = bias ? bias[F] : 0.f;
#pragma unroll
        for (int r = 0; r < 8; ++r) {
            float z = acc[t][r] + bv;
            if (doAct) z = silu_f(z);
            stg[(r + 8 * h) * DIM + F] = (__bf16)z;
        }
    }
}

// Row-major bf16 staging -> A fragments (lane = fixed row, two b128 loads per chunk)
static __device__ __forceinline__ void load_afrags(const __bf16* stg, int lane, Frag* A) {
    int nl = lane & 15, h = lane >> 4;
#pragma unroll
    for (int c = 0; c < 4; ++c) {
#pragma unroll
        for (int run = 0; run < 2; ++run) {
            int f0 = c * 32 + run * 16 + h * 8;
            const uint4* p = (const uint4*)(stg + nl * DIM + f0);
            uint4 q = *p;
            A[c].u[run*4+0]=q.x; A[c].u[run*4+1]=q.y;
            A[c].u[run*4+2]=q.z; A[c].u[run*4+3]=q.w;
        }
    }
}

static __device__ __forceinline__ void store_out(const v8f* acc, float* __restrict__ out,
                                                 int rowbase, int lane) {
    int nl = lane & 15, h = lane >> 4;
#pragma unroll
    for (int t = 0; t < 8; ++t) {
        int F = t * 16 + nl;
#pragma unroll
        for (int r = 0; r < 8; ++r) {
            out[(size_t)(rowbase + r + 8 * h) * DIM + F] = acc[t][r];
        }
    }
}

// Shared MLP tail: 3x(Linear+bias+SiLU) + Linear, two row-tiles per wave.
static __device__ __forceinline__ void mlp_tail(Frag* A0, Frag* A1,
                                                const uint32_t* wl, const float* bias_l,
                                                __bf16* stg, int lane,
                                                float* __restrict__ out, int rbase) {
    v8f acc0[8], acc1[8];
#pragma unroll 1
    for (int L = 0; L < 3; ++L) {
#pragma unroll
        for (int t = 0; t < 8; ++t) { acc0[t] = vzero(); acc1[t] = vzero(); }
        gemm_layer2(A0, A1, wl + L * 8192, lane, acc0, acc1);
        act_store(acc0, bias_l + L * DIM, stg, lane, true);
        LDS_FENCE();
        load_afrags(stg, lane, A0);
        LDS_FENCE();
        act_store(acc1, bias_l + L * DIM, stg, lane, true);
        LDS_FENCE();
        load_afrags(stg, lane, A1);
        LDS_FENCE();
    }
#pragma unroll
    for (int t = 0; t < 8; ++t) { acc0[t] = vzero(); acc1[t] = vzero(); }
    gemm_layer2(A0, A1, wl + 3 * 8192, lane, acc0, acc1);
    store_out(acc0, out, rbase, lane);
    store_out(acc1, out, rbase + 16, lane);
}

// ---------------------------------------------------------------- edge branch
static __device__ __forceinline__ void build_edge_x0(const float* __restrict__ m_ji,
                                                     const float* __restrict__ e_rbf,
                                                     const float* we_l,
                                                     int row, int h, Frag* A) {
    float rbf[NRBF];
#pragma unroll
    for (int j = 0; j < NRBF; ++j) rbf[j] = e_rbf[(size_t)row * NRBF + j];
#pragma unroll
    for (int c = 0; c < 4; ++c) {
#pragma unroll
        for (int run = 0; run < 2; ++run) {
            int f0 = c * 32 + run * 16 + h * 8;
            const float4* mp = (const float4*)(m_ji + (size_t)row * DIM + f0);
            float4 m0 = mp[0], m1 = mp[1];
            float mv[8] = {m0.x,m0.y,m0.z,m0.w,m1.x,m1.y,m1.z,m1.w};
            float x[8];
#pragma unroll
            for (int k = 0; k < 8; ++k) {
                float p = 0.f;
#pragma unroll
                for (int j = 0; j < NRBF; ++j) p += rbf[j] * we_l[j * DIM + f0 + k];
                x[k] = p * mv[k];
            }
#pragma unroll
            for (int q = 0; q < 4; ++q) A[c].u[run*4+q] = pk2(x[2*q], x[2*q+1]);
        }
    }
}

__global__ void __launch_bounds__(128) dime_edge_kernel(
    const float* __restrict__ m_ji, const float* __restrict__ e_rbf,
    const float* __restrict__ We,
    const float* __restrict__ W1, const float* __restrict__ b1,
    const float* __restrict__ W2, const float* __restrict__ b2,
    const float* __restrict__ W3, const float* __restrict__ b3,
    const float* __restrict__ W4,
    float* __restrict__ out)
{
    extern __shared__ unsigned char smem[];
    uint32_t* wl      = (uint32_t*)smem;                       // 4 * 32 KB
    float*    we_l    = (float*)(smem + 131072);               // 6*128 f32
    float*    bias_l  = (float*)(smem + 131072 + 3072);        // 3*128 f32
    __bf16*   stg_all = (__bf16*)(smem + 131072 + 3072 + 1536);// 4 * 4 KB

    int tid = threadIdx.x, nthr = blockDim.x;
    stage_wchunks(wl + 0*8192, W1, DIM, 4, tid, nthr);
    stage_wchunks(wl + 1*8192, W2, DIM, 4, tid, nthr);
    stage_wchunks(wl + 2*8192, W3, DIM, 4, tid, nthr);
    stage_wchunks(wl + 3*8192, W4, DIM, 4, tid, nthr);
    for (int i = tid; i < NRBF * DIM; i += nthr) we_l[i] = We[i];
    for (int i = tid; i < DIM; i += nthr) {
        bias_l[i] = b1[i]; bias_l[DIM + i] = b2[i]; bias_l[2*DIM + i] = b3[i];
    }
    __syncthreads();

    int lane = tid & 31, wave = tid >> 5;
    int nl = lane & 15, h = lane >> 4;
    __bf16* stg = stg_all + wave * (16 * DIM);

    int ngroups = E_EDGES / 128;          // 128 rows per WG iteration
    for (int g = blockIdx.x; g < ngroups; g += gridDim.x) {
        int rbase = g * 128 + wave * 32;

        // prefetch next iteration's streaming rows (global_prefetch_b8)
        int gn = g + gridDim.x;
        if (gn < ngroups) {
            int nrb = gn * 128 + wave * 32;
            __builtin_prefetch(m_ji + (size_t)(nrb + nl) * DIM + h * 64, 0, 0);
            __builtin_prefetch(m_ji + (size_t)(nrb + 16 + nl) * DIM + h * 64, 0, 0);
            __builtin_prefetch(e_rbf + (size_t)(nrb + nl * 2 + h) * NRBF, 0, 0);
        }

        Frag A0[4], A1[4];
        build_edge_x0(m_ji, e_rbf, we_l, rbase + nl,      h, A0);
        build_edge_x0(m_ji, e_rbf, we_l, rbase + 16 + nl, h, A1);
        mlp_tail(A0, A1, wl, bias_l, stg, lane, out, rbase);
    }
}

// --------------------------------------------------------------- angle branch
static __device__ __forceinline__ Frag build_sbf_frag(const float* __restrict__ a_sbf,
                                                      int row, int h) {
    Frag As;
#pragma unroll
    for (int run = 0; run < 2; ++run) {
        int f0 = run * 16 + h * 8;
        float x[8];
#pragma unroll
        for (int k = 0; k < 8; ++k) x[k] = a_sbf[(size_t)row * NSBF + f0 + k];
#pragma unroll
        for (int q = 0; q < 4; ++q) As.u[run*4+q] = pk2(x[2*q], x[2*q+1]);
    }
    return As;
}

// x0 = (proj_readback + sbf_tail @ Wa_tail) * (m_ji[kj] + m_ji[ji]), A-frag layout
static __device__ __forceinline__ void build_angle_x0(const __bf16* stg,
                                                      const float* __restrict__ m_ji,
                                                      const float* __restrict__ a_sbf,
                                                      const float* wa_tail,
                                                      int kj, int ji, int row,
                                                      int nl, int h, Frag* A) {
    float tl[10];
#pragma unroll
    for (int j = 0; j < 10; ++j) tl[j] = a_sbf[(size_t)row * NSBF + 32 + j];
    const float* mkj = m_ji + (size_t)kj * DIM;
    const float* mji = m_ji + (size_t)ji * DIM;
#pragma unroll
    for (int c = 0; c < 4; ++c) {
#pragma unroll
        for (int run = 0; run < 2; ++run) {
            int f0 = c * 32 + run * 16 + h * 8;
            const uint4* pp = (const uint4*)(stg + nl * DIM + f0);
            uint4 q = *pp;
            uint32_t pu[4] = {q.x, q.y, q.z, q.w};
            const float4* g1 = (const float4*)(mkj + f0);
            const float4* g2 = (const float4*)(mji + f0);
            float4 ga0 = g1[0], ga1 = g1[1], gb0 = g2[0], gb1 = g2[1];
            float gv[8] = {ga0.x+gb0.x, ga0.y+gb0.y, ga0.z+gb0.z, ga0.w+gb0.w,
                           ga1.x+gb1.x, ga1.y+gb1.y, ga1.z+gb1.z, ga1.w+gb1.w};
            float x[8];
#pragma unroll
            for (int k = 0; k < 8; ++k) {
                uint16_t pb = (k & 1) ? (uint16_t)(pu[k >> 1] >> 16)
                                      : (uint16_t)(pu[k >> 1] & 0xffff);
                float p = bf2f(pb);
#pragma unroll
                for (int j = 0; j < 10; ++j) p += tl[j] * wa_tail[j * DIM + f0 + k];
                x[k] = p * gv[k];
            }
#pragma unroll
            for (int qi = 0; qi < 4; ++qi) A[c].u[run*4+qi] = pk2(x[2*qi], x[2*qi+1]);
        }
    }
}

__global__ void __launch_bounds__(128) dime_angle_kernel(
    const float* __restrict__ m_ji, const float* __restrict__ a_sbf,
    const int* __restrict__ kj_idx, const int* __restrict__ ji_idx,
    const float* __restrict__ Wa,
    const float* __restrict__ W1, const float* __restrict__ b1,
    const float* __restrict__ W2, const float* __restrict__ b2,
    const float* __restrict__ W3, const float* __restrict__ b3,
    const float* __restrict__ W4,
    float* __restrict__ out)
{
    extern __shared__ unsigned char smem[];
    uint32_t* wl      = (uint32_t*)smem;                                 // 128 KB
    uint32_t* wa0     = (uint32_t*)(smem + 131072);                      // chunk0 B-frag, 8 KB
    float*    wa_tail = (float*)(smem + 131072 + 8192);                  // rows 32..41, 5 KB
    float*    bias_l  = (float*)(smem + 131072 + 8192 + 5120);           // 1.5 KB
    __bf16*   stg_all = (__bf16*)(smem + 131072 + 8192 + 5120 + 1536);   // 16 KB

    int tid = threadIdx.x, nthr = blockDim.x;
    stage_wchunks(wl + 0*8192, W1, DIM, 4, tid, nthr);
    stage_wchunks(wl + 1*8192, W2, DIM, 4, tid, nthr);
    stage_wchunks(wl + 2*8192, W3, DIM, 4, tid, nthr);
    stage_wchunks(wl + 3*8192, W4, DIM, 4, tid, nthr);
    stage_wchunks(wa0, Wa, NSBF, 1, tid, nthr);          // K = 0..31 only
    for (int i = tid; i < 10 * DIM; i += nthr)
        wa_tail[i] = Wa[(size_t)(32 + i / DIM) * DIM + (i % DIM)];
    for (int i = tid; i < DIM; i += nthr) {
        bias_l[i] = b1[i]; bias_l[DIM + i] = b2[i]; bias_l[2*DIM + i] = b3[i];
    }
    __syncthreads();

    int lane = tid & 31, wave = tid >> 5;
    int nl = lane & 15, h = lane >> 4;
    __bf16* stg = stg_all + wave * (16 * DIM);

    int ngroups = T_TRIP / 128;
    for (int g = blockIdx.x; g < ngroups; g += gridDim.x) {
        int rbase = g * 128 + wave * 32;
        int row0 = rbase + nl;
        int row1 = rbase + 16 + nl;

        int gn = g + gridDim.x;
        if (gn < ngroups) {
            int nrb = gn * 128 + wave * 32;
            __builtin_prefetch(a_sbf + (size_t)(nrb + nl * 2 + h) * NSBF, 0, 0);
            __builtin_prefetch(kj_idx + nrb + lane, 0, 0);
            __builtin_prefetch(ji_idx + nrb + lane, 0, 0);
        }

        int kj0 = kj_idx[row0], ji0 = ji_idx[row0];
        int kj1 = kj_idx[row1], ji1 = ji_idx[row1];

        // projection (K chunk 0 of 42) for both tiles, sharing pipelined Wa B fragments
        Frag As0 = build_sbf_frag(a_sbf, row0, h);
        Frag As1 = build_sbf_frag(a_sbf, row1, h);
        v8f acc0[8], acc1[8];
#pragma unroll
        for (int t = 0; t < 8; ++t) { acc0[t] = vzero(); acc1[t] = vzero(); }
        {
            v16bf a0 = frag_cast(As0);
            v16bf a1 = frag_cast(As1);
            v16bf b = load_bfrag(wa0, 0, lane);
#pragma unroll
            for (int t = 0; t < 8; ++t) {
                v16bf bn = (t < 7) ? load_bfrag(wa0, t + 1, lane) : b;
                acc0[t] = __builtin_amdgcn_wmma_f32_16x16x32_bf16(
                    false, a0, false, b, (short)0, acc0[t], false, false);
                acc1[t] = __builtin_amdgcn_wmma_f32_16x16x32_bf16(
                    false, a1, false, b, (short)0, acc1[t], false, false);
                b = bn;
            }
        }

        Frag A0[4], A1[4];
        act_store(acc0, nullptr, stg, lane, false);
        LDS_FENCE();
        build_angle_x0(stg, m_ji, a_sbf, wa_tail, kj0, ji0, row0, nl, h, A0);
        LDS_FENCE();
        act_store(acc1, nullptr, stg, lane, false);
        LDS_FENCE();
        build_angle_x0(stg, m_ji, a_sbf, wa_tail, kj1, ji1, row1, nl, h, A1);
        LDS_FENCE();

        mlp_tail(A0, A1, wl, bias_l, stg, lane, out, rbase);
    }
}

extern "C" void kernel_launch(void* const* d_in, const int* in_sizes, int n_in,
                              void* d_out, int out_size, void* d_ws, size_t ws_size,
                              hipStream_t stream) {
    (void)in_sizes; (void)n_in; (void)out_size; (void)d_ws; (void)ws_size;
    const float* m_ji  = (const float*)d_in[0];
    const float* e_rbf = (const float*)d_in[1];
    const float* a_sbf = (const float*)d_in[2];
    // d_in[3] nbr_list unused
    const int* kj_idx  = (const int*)d_in[4];
    const int* ji_idx  = (const int*)d_in[5];
    const float* We    = (const float*)d_in[6];
    const float* W_e1  = (const float*)d_in[7];
    const float* b_e1  = (const float*)d_in[8];
    const float* W_e2  = (const float*)d_in[9];
    const float* b_e2  = (const float*)d_in[10];
    const float* W_e3  = (const float*)d_in[11];
    const float* b_e3  = (const float*)d_in[12];
    const float* W_e4  = (const float*)d_in[13];
    const float* Wa    = (const float*)d_in[14];
    const float* W_a1  = (const float*)d_in[15];
    const float* b_a1  = (const float*)d_in[16];
    const float* W_a2  = (const float*)d_in[17];
    const float* b_a2  = (const float*)d_in[18];
    const float* W_a3  = (const float*)d_in[19];
    const float* b_a3  = (const float*)d_in[20];
    const float* W_a4  = (const float*)d_in[21];

    float* out_edge  = (float*)d_out;
    float* out_angle = out_edge + (size_t)E_EDGES * DIM;

    size_t smem_edge  = 131072 + 3072 + 1536 + 4 * 16 * DIM * sizeof(__bf16);          // ~148.5 KB
    size_t smem_angle = 131072 + 8192 + 5120 + 1536 + 4 * 16 * DIM * sizeof(__bf16);   // ~158.5 KB

    dime_edge_kernel<<<1024, 128, smem_edge, stream>>>(
        m_ji, e_rbf, We, W_e1, b_e1, W_e2, b_e2, W_e3, b_e3, W_e4, out_edge);
    dime_angle_kernel<<<2048, 128, smem_angle, stream>>>(
        m_ji, a_sbf, kj_idx, ji_idx, Wa, W_a1, b_a1, W_a2, b_a2, W_a3, b_a3, W_a4, out_angle);
}
